// E3DLSTM_69982197121320
// MI455X (gfx1250) — matrix operand order repaired and checked by reading, not compiled, not run on hardware
//
#include <hip/hip_runtime.h>

// ---------------------------------------------------------------------------
// E3D-LSTM forward for gfx1250 (MI455X). All GEMM-like reductions (5x5 convs
// as implicit GEMM, 1x1 mixing conv, eidetic attention, final projection) run
// through v_wmma_f32_16x16x32_f16 (f16 in, f32 accumulate). Working set
// (<50MB) is L2-resident (192MB), so the kernel set optimizes for matrix-pipe
// throughput and launch-chain depth, not HBM bandwidth. Conv waves compute
// 32x16 output tiles so one im2col B-fragment build feeds two WMMAs.
// ---------------------------------------------------------------------------

typedef __attribute__((ext_vector_type(16))) _Float16 v16h;
typedef __attribute__((ext_vector_type(8)))  float    v8f;

__device__ __forceinline__ float sigf(float x) { return 1.0f / (1.0f + __expf(-x)); }

// ---- weight prep: f32 (Cout,K) row-major -> f16 (Cout,Kpad) zero-padded ----
__global__ void k_prepw(const float* __restrict__ src, _Float16* __restrict__ dst,
                        int Cout, int K, int Kpad) {
  int tid = blockIdx.x * blockDim.x + threadIdx.x;
  int n = Cout * Kpad;
  if (tid >= n) return;
  int co = tid / Kpad;
  int k  = tid - co * Kpad;
  dst[tid] = (k < K) ? (_Float16)src[(size_t)co * K + k] : (_Float16)0.0f;
}

// ---- f32 (8,C,512) -> f16 (8,dstCB,512) at channel offset coff -------------
__global__ void k_convert(const float* __restrict__ src, _Float16* __restrict__ dst,
                          int C, int dstCB, int coff) {
  int tid = blockIdx.x * blockDim.x + threadIdx.x;
  int n = 8 * C * 512;
  if (tid >= n) return;
  int cs = C << 9;
  int b = tid / cs;
  int r = tid - b * cs;
  int c = r >> 9;
  int p = r & 511;
  dst[((size_t)(b * dstCB + coff + c) << 9) + p] = (_Float16)src[tid];
}

// ---- frame scheduler: shift window (d1 -> d0) and build new d1 slice -------
// ci16 layout: (b, c=16, d=2, 256) f16. t<5: raw frame; else masked blend.
__global__ void k_frame_in(const float* __restrict__ frames, const float* __restrict__ mask,
                           const float* __restrict__ gen, _Float16* __restrict__ ci16, int t) {
  int tid = blockIdx.x * blockDim.x + threadIdx.x;  // 8*16*256 = 32768
  if (tid >= 32768) return;
  int b = tid >> 12;
  int c = (tid >> 8) & 15;
  int p = tid & 255;
  _Float16* base = ci16 + ((size_t)(b * 16 + c) << 9);
  _Float16 oldv = base[256 + p];
  float f = frames[(((size_t)(b * 10 + t) * 16 + c) << 8) + p];
  float v;
  if (t < 5) {
    v = f;
  } else {
    float m = mask[(((size_t)(b * 4 + (t - 5)) * 16 + c) << 8) + p];
    v = m * f + (1.0f - m) * gen[(((size_t)b * 16 + c) << 8) + p];
  }
  base[p] = oldv;
  base[256 + p] = (_Float16)v;
}

// ---- append previous c1 into history slot (b,640,512 buffer) ---------------
__global__ void k_hist(const float* __restrict__ c1, float* __restrict__ hist, int slot) {
  int tid = blockIdx.x * blockDim.x + threadIdx.x;  // 262144
  if (tid >= 262144) return;
  int b = tid >> 15;
  int r = tid & 32767;
  hist[(size_t)b * 327680 + (size_t)slot * 32768 + r] = c1[tid];
}

// ---------------------------------------------------------------------------
// Generic implicit-GEMM conv, one wave32 per 32(co) x 16(w-row) output tile.
// N tile = one image row: blockIdx.x -> (b,d,h), lane&15 -> w.
// A (weights, f16 A-layout): lane l row M=l&15, k = k0 + (j>>3)*16 + half*8 + (j&7)
// B (im2col, f16 B-layout):  lane l col N=l&15, k = k0 + (j>>3)*16 + half*8 + (j&7)
// One B-fragment build (16 guarded loads + addr math) feeds two WMMAs against
// two 16-row weight tiles; next-chunk weight/input prefetch hides the serial
// load->cvt->wmma chain. TAPS=25: 5x5 spatial taps; TAPS=1: pointwise.
// ---------------------------------------------------------------------------
template <int TAPS>
__global__ void k_conv_wmma(const _Float16* __restrict__ in16, const _Float16* __restrict__ w16,
                            const float* __restrict__ bias, float* __restrict__ out,
                            int Cin, int Cout, int Kpad, int inCB) {
  const int lane = threadIdx.x;
  const int nt = blockIdx.x;            // 0..255
  const int mt = blockIdx.y;            // 32-channel output tile
  const int b  = nt >> 5;
  const int d  = (nt >> 4) & 1;
  const int h  = nt & 15;
  const int wq = lane & 15;
  const int hf = lane >> 4;
  const int K  = Cin * TAPS;

  const _Float16* wrow0 = w16 + (size_t)(mt * 32 + wq) * Kpad;
  const _Float16* wrow1 = wrow0 + ((size_t)16 * Kpad);
  const _Float16* inb   = in16 + ((size_t)b * inCB << 9);

  v8f acc0 = {}, acc1 = {};
  for (int k0 = 0; k0 < Kpad; k0 += 32) {
    // prefetch next chunk (speculative; lowers to global_prefetch_b8)
    if (k0 + 32 < Kpad) {
      __builtin_prefetch(wrow0 + k0 + 32, 0, 0);
      __builtin_prefetch(wrow1 + k0 + 32, 0, 0);
      int kn = k0 + 32 + (hf << 4);
      int cin_n = (TAPS == 25) ? (kn / 25) : kn;
      if (cin_n >= Cin) cin_n = Cin - 1;
      __builtin_prefetch(inb + (((size_t)cin_n << 9) + (d << 8) + (h << 4) + wq), 0, 0);
    }
    v16h a0, a1, bb;
#pragma unroll
    for (int j = 0; j < 16; ++j) {
      int ka = k0 + ((j >> 3) << 4) + (hf << 3) + (j & 7);
      a0[j] = wrow0[ka];
      a1[j] = wrow1[ka];
    }
#pragma unroll
    for (int j = 0; j < 16; ++j) {
      int k = k0 + ((j >> 3) << 4) + (hf << 3) + (j & 7);
      _Float16 v = (_Float16)0.0f;
      if (k < K) {
        int ci, hh, ww;
        if (TAPS == 25) {
          ci = k / 25;
          int r  = k - ci * 25;
          int kh = r / 5;
          int kw = r - kh * 5;
          hh = h + kh - 2;
          ww = wq + kw - 2;
        } else {
          ci = k; hh = h; ww = wq;
        }
        if ((unsigned)hh < 16u && (unsigned)ww < 16u)
          v = inb[((size_t)ci << 9) + (d << 8) + (hh << 4) + ww];
      }
      bb[j] = v;
    }
    acc0 = __builtin_amdgcn_wmma_f32_16x16x32_f16(false, a0, false, bb, (short)0, acc0, false, false);
    acc1 = __builtin_amdgcn_wmma_f32_16x16x32_f16(false, a1, false, bb, (short)0, acc1, false, false);
  }
#pragma unroll
  for (int r = 0; r < 8; ++r) {
    int co0 = mt * 32 + r + (hf << 3);  // C/D layout: M = r + 8*half, N = lane&15
    int co1 = co0 + 16;
    out[(((size_t)(b * Cout + co0) * 2 + d) << 8) + (h << 4) + wq] = acc0[r] + bias[co0];
    out[(((size_t)(b * Cout + co1) * 2 + d) << 8) + (h << 4) + wq] = acc1[r] + bias[co1];
  }
}

// ---- per-sample layernorm over C*512 elements, then per-element gain/bias --
__global__ void k_layernorm(float* __restrict__ x, const float* __restrict__ g,
                            const float* __restrict__ bt, int C) {
  __shared__ float rs[256], rq[256];
  int b = blockIdx.x;
  int n = C << 9;
  float* xb = x + (size_t)b * n;
  float s = 0.f, q = 0.f;
  for (int i = threadIdx.x; i < n; i += 256) { float v = xb[i]; s += v; q += v * v; }
  rs[threadIdx.x] = s; rq[threadIdx.x] = q;
  __syncthreads();
  for (int o = 128; o > 0; o >>= 1) {
    if ((int)threadIdx.x < o) { rs[threadIdx.x] += rs[threadIdx.x + o]; rq[threadIdx.x] += rq[threadIdx.x + o]; }
    __syncthreads();
  }
  float inv_n = 1.0f / (float)n;
  float mean = rs[0] * inv_n;
  float var  = rq[0] * inv_n - mean * mean;
  float rinv = rsqrtf(var + 1e-5f);
  for (int i = threadIdx.x; i < n; i += 256)
    xb[i] = (xb[i] - mean) * rinv * g[i] + bt[i];
}

// ---- gate kernels ----------------------------------------------------------
__global__ void k_gates1(const float* __restrict__ xcat, const float* __restrict__ hcat,
                         float* __restrict__ iT, float* __restrict__ gT, float* __restrict__ rT) {
  int tid = blockIdx.x * blockDim.x + threadIdx.x;
  if (tid >= 262144) return;
  int b = tid >> 15, r = tid & 32767;
  const float* xb = xcat + (size_t)b * 229376;
  const float* hb = hcat + (size_t)b * 131072;
  iT[tid] = sigf(xb[r] + hb[r]);
  gT[tid] = tanhf(xb[32768 + r] + hb[32768 + r]);
  rT[tid] = sigf(xb[65536 + r] + hb[65536 + r]);
}

__global__ void k_gates2(const float* __restrict__ xcat, const float* __restrict__ gcat,
                         float* __restrict__ gm) {
  int tid = blockIdx.x * blockDim.x + threadIdx.x;
  if (tid >= 262144) return;
  int b = tid >> 15, r = tid & 32767;
  const float* xb = xcat + (size_t)b * 229376;
  const float* gb = gcat + (size_t)b * 131072;
  float ti = sigf(xb[131072 + r] + gb[r]);                  // ti_x + i_m
  float tf = sigf(xb[196608 + r] + gb[32768 + r] + 1.0f);   // tf_x + f_m + FORGET_BIAS
  float tg = tanhf(xb[163840 + r] + gb[65536 + r]);         // tg_x + g_m
  float mm = gb[98304 + r];                                 // m_m
  gm[tid] = tf * tanhf(mm) + ti * tg;
}

__global__ void k_add2(const float* __restrict__ a, const float* __restrict__ b2,
                       float* __restrict__ o) {
  int tid = blockIdx.x * blockDim.x + threadIdx.x;
  if (tid >= 262144) return;
  o[tid] = a[tid] + b2[tid];
}

__global__ void k_cnew(const float* __restrict__ iT, const float* __restrict__ gT,
                       const float* __restrict__ cn, float* __restrict__ c) {
  int tid = blockIdx.x * blockDim.x + threadIdx.x;
  if (tid >= 262144) return;
  c[tid] = iT[tid] * gT[tid] + cn[tid];
}

__global__ void k_outgate(const float* __restrict__ xcat, const float* __restrict__ hcat,
                          const float* __restrict__ oc, const float* __restrict__ om,
                          const float* __restrict__ memb, float* __restrict__ h) {
  int tid = blockIdx.x * blockDim.x + threadIdx.x;
  if (tid >= 262144) return;
  int b = tid >> 15, r = tid & 32767;
  float ot = tanhf(xcat[(size_t)b * 229376 + 98304 + r] + hcat[(size_t)b * 131072 + 98304 + r] +
                   oc[tid] + om[tid]);
  h[tid] = memb[tid] * tanhf(ot);
}

// ---------------------------------------------------------------------------
// Fused flash-style eidetic attention. One wave32 per (batch, 16 Q rows).
// qf[x,c] = q_flat[b][x*64+c]; kf[y,c] = kv_flat[b][y*64+c] (valid prefix Y).
// Streams 32 keys/iter: 4 WMMAs QK^T (K-dim=64 split in 2, two 16-col subtiles),
// online softmax via 16-lane shfl_xor (C-layout rows live in 16-lane halves),
// LDS repack of P into A-layout, 4 WMMAs for PV (K-dim = 32 keys).
// ---------------------------------------------------------------------------
__global__ void k_attn(const float* __restrict__ q, const float* __restrict__ kv,
                       float* __restrict__ out, int Y, int kvStride) {
  const int lane = threadIdx.x;
  const int b  = blockIdx.y;
  const int hf = lane >> 4;
  const int nl = lane & 15;
  const float* qb = q  + ((size_t)b << 15);
  const float* kb = kv + (size_t)b * kvStride;
  float* ob = out + ((size_t)b << 15);
  const int xr = blockIdx.x * 16 + nl;

  __shared__ _Float16 pl[512];   // 16 rows x 32 keys

  v16h qa0, qa1;
#pragma unroll
  for (int j = 0; j < 16; ++j) {
    int cl = ((j >> 3) << 4) + (hf << 3) + (j & 7);
    qa0[j] = (_Float16)qb[(xr << 6) + cl];
    qa1[j] = (_Float16)qb[(xr << 6) + 32 + cl];
  }

  float mrun[8], lrun[8];
  v8f acc[4] = {};
#pragma unroll
  for (int r = 0; r < 8; ++r) { mrun[r] = -1e30f; lrun[r] = 0.0f; }

  for (int yb = 0; yb < Y; yb += 32) {
    v8f s0 = {}, s1 = {};
    {
      const int y = yb + nl;
      v16h kf;
#pragma unroll
      for (int j = 0; j < 16; ++j) kf[j] = (_Float16)kb[(y << 6) + ((j >> 3) << 4) + (hf << 3) + (j & 7)];
      s0 = __builtin_amdgcn_wmma_f32_16x16x32_f16(false, qa0, false, kf, (short)0, s0, false, false);
#pragma unroll
      for (int j = 0; j < 16; ++j) kf[j] = (_Float16)kb[(y << 6) + 32 + ((j >> 3) << 4) + (hf << 3) + (j & 7)];
      s0 = __builtin_amdgcn_wmma_f32_16x16x32_f16(false, qa1, false, kf, (short)0, s0, false, false);
    }
    {
      const int y = yb + 16 + nl;
      v16h kf;
#pragma unroll
      for (int j = 0; j < 16; ++j) kf[j] = (_Float16)kb[(y << 6) + ((j >> 3) << 4) + (hf << 3) + (j & 7)];
      s1 = __builtin_amdgcn_wmma_f32_16x16x32_f16(false, qa0, false, kf, (short)0, s1, false, false);
#pragma unroll
      for (int j = 0; j < 16; ++j) kf[j] = (_Float16)kb[(y << 6) + 32 + ((j >> 3) << 4) + (hf << 3) + (j & 7)];
      s1 = __builtin_amdgcn_wmma_f32_16x16x32_f16(false, qa1, false, kf, (short)0, s1, false, false);
    }

#pragma unroll
    for (int r = 0; r < 8; ++r) {
      float tmax = fmaxf(s0[r], s1[r]);
#pragma unroll
      for (int mk = 8; mk >= 1; mk >>= 1) tmax = fmaxf(tmax, __shfl_xor(tmax, mk, 32));
      float mnew = fmaxf(mrun[r], tmax);
      float sc = __expf(mrun[r] - mnew);
      float p0 = __expf(s0[r] - mnew);
      float p1 = __expf(s1[r] - mnew);
      float ps = p0 + p1;
#pragma unroll
      for (int mk = 8; mk >= 1; mk >>= 1) ps += __shfl_xor(ps, mk, 32);
      lrun[r] = lrun[r] * sc + ps;
      mrun[r] = mnew;
#pragma unroll
      for (int cf = 0; cf < 4; ++cf) acc[cf][r] = acc[cf][r] * sc;
      int M = r + (hf << 3);
      pl[M * 32 + nl]      = (_Float16)p0;
      pl[M * 32 + 16 + nl] = (_Float16)p1;
    }
    __syncthreads();
    v16h pa;
#pragma unroll
    for (int j = 0; j < 16; ++j) {
      int kl = ((j >> 3) << 4) + (hf << 3) + (j & 7);
      pa[j] = pl[nl * 32 + kl];
    }
#pragma unroll
    for (int cf = 0; cf < 4; ++cf) {
      v16h vf;
#pragma unroll
      for (int j = 0; j < 16; ++j) {
        int y = yb + ((j >> 3) << 4) + (hf << 3) + (j & 7);
        vf[j] = (_Float16)kb[(y << 6) + (cf << 4) + nl];
      }
      acc[cf] = __builtin_amdgcn_wmma_f32_16x16x32_f16(false, pa, false, vf, (short)0, acc[cf], false, false);
    }
    __syncthreads();
  }

#pragma unroll
  for (int r = 0; r < 8; ++r) {
    float inv = 1.0f / lrun[r];
    int x = blockIdx.x * 16 + r + (hf << 3);
#pragma unroll
    for (int cf = 0; cf < 4; ++cf)
      ob[(x << 6) + (cf << 4) + nl] = acc[cf][r] * inv;
  }
}

// ---- final projection: gen = conv(h1, w_last, stride D) via WMMA -----------
// M=16 (frame channels), K=128 (ci*2+kd), N = b*256+hw. One wave per 16 cols.
__global__ void k_gen(const _Float16* __restrict__ h16, const _Float16* __restrict__ w16,
                      float* __restrict__ genb, float* __restrict__ out2, int slot) {
  const int lane = threadIdx.x;
  const int nt = blockIdx.x;           // 0..127
  const int b  = nt >> 4;
  const int pos = ((nt & 15) << 4) + (lane & 15);
  const int hf = lane >> 4;
  const _Float16* wrow = w16 + ((size_t)(lane & 15) << 7);
  const _Float16* hb = h16 + ((size_t)b << 15);
  v8f acc = {};
  for (int k0 = 0; k0 < 128; k0 += 32) {
    v16h a, bb;
#pragma unroll
    for (int j = 0; j < 16; ++j)
      a[j] = wrow[k0 + ((j >> 3) << 4) + (hf << 3) + (j & 7)];
#pragma unroll
    for (int j = 0; j < 16; ++j) {
      int k = k0 + ((j >> 3) << 4) + (hf << 3) + (j & 7);
      int ci = k >> 1, kd = k & 1;
      bb[j] = hb[(ci << 9) + (kd << 8) + pos];
    }
    acc = __builtin_amdgcn_wmma_f32_16x16x32_f16(false, a, false, bb, (short)0, acc, false, false);
  }
#pragma unroll
  for (int r = 0; r < 8; ++r) {
    int co = r + (hf << 3);
    float v = acc[r];
    genb[(((size_t)b * 16 + co) << 8) + pos] = v;
    if (out2) out2[(((size_t)(b * 5 + slot) * 16 + co) << 8) + pos] = v;
  }
}

// ---------------------------------------------------------------------------
// Host orchestration. Input leaf order follows JAX's sorted-dict flattening:
//   0: frames_tensor, 1: mask_true,
//   2..25: cells[0] leaves (sorted keys), 26..49: cells[1], 50: w_last.
// Sorted cell keys: b_cell_last,bh,bm,bnc,bng,bx, ln_c_b,ln_c_g, ln_h_b,ln_h_g,
//   ln_m_b,ln_m_g, ln_nc_b,ln_nc_g, ln_ng_b,ln_ng_g, ln_x_b,ln_x_g,
//   w_cell_last, wh, wm, wnc, wng, wx.
// ---------------------------------------------------------------------------
extern "C" void kernel_launch(void* const* d_in, const int* in_sizes, int n_in,
                              void* d_out, int out_size, void* d_ws, size_t ws_size,
                              hipStream_t stream) {
  (void)in_sizes; (void)n_in; (void)out_size; (void)ws_size;
  const float* frames = (const float*)d_in[0];
  const float* maskp  = (const float*)d_in[1];
  auto P = [&](int i) { return (const float*)d_in[i]; };
  const int cbase[2] = {2, 26};

  // ---- workspace carve-up (~46 MB) ----
  char* wsp = (char*)d_ws;
  auto AF = [&](size_t n) { float* p = (float*)wsp; wsp += n * sizeof(float); return p; };
  auto AH = [&](size_t n) { _Float16* p = (_Float16*)wsp; wsp += n * sizeof(_Float16); return p; };

  float* h0 = AF(262144); float* h1 = AF(262144);
  float* c0 = AF(262144); float* c1 = AF(262144);
  float* gmem = AF(262144);
  float* iT = AF(262144); float* gT = AF(262144); float* rT = AF(262144);
  float* recall = AF(262144); float* cn = AF(262144);
  float* oc = AF(262144); float* om = AF(262144); float* memb = AF(262144);
  float* xcat = AF(1835008); float* hcat = AF(1048576); float* gcat = AF(1048576);
  float* hist1 = AF(2621440); float* genb = AF(32768);
  _Float16* in16 = AH(524288);
  _Float16* ci16 = AH(65536);
  _Float16* wx16[2]  = { AH(448 * 416), AH(448 * 1600) };
  _Float16* wh16[2]  = { AH(409600), AH(409600) };
  _Float16* wm16[2]  = { AH(409600), AH(409600) };
  _Float16* wnc16[2] = { AH(102400), AH(102400) };
  _Float16* wng16[2] = { AH(102400), AH(102400) };
  _Float16* wcl16[2] = { AH(8192), AH(8192) };
  _Float16* wlast16  = AH(2048);

  // ---- zero recurrent state (graph-capturable memsets) ----
  hipMemsetAsync(h0, 0, 262144 * sizeof(float), stream);
  hipMemsetAsync(h1, 0, 262144 * sizeof(float), stream);
  hipMemsetAsync(c0, 0, 262144 * sizeof(float), stream);
  hipMemsetAsync(c1, 0, 262144 * sizeof(float), stream);
  hipMemsetAsync(gmem, 0, 262144 * sizeof(float), stream);
  hipMemsetAsync(genb, 0, 32768 * sizeof(float), stream);
  hipMemsetAsync(hist1, 0, 2621440 * sizeof(float), stream);
  hipMemsetAsync(ci16, 0, 65536 * sizeof(_Float16), stream);

  // ---- weight repack to f16, K padded to multiple of 32 ----
  auto prep = [&](const float* s, _Float16* dp, int Cout, int K, int Kpad) {
    int n = Cout * Kpad;
    k_prepw<<<(n + 255) / 256, 256, 0, stream>>>(s, dp, Cout, K, Kpad);
  };
  prep(P(cbase[0] + 23), wx16[0], 448, 400, 416);
  prep(P(cbase[1] + 23), wx16[1], 448, 1600, 1600);
  for (int L = 0; L < 2; ++L) {
    prep(P(cbase[L] + 19), wh16[L], 256, 1600, 1600);
    prep(P(cbase[L] + 20), wm16[L], 256, 1600, 1600);
    prep(P(cbase[L] + 21), wnc16[L], 64, 1600, 1600);
    prep(P(cbase[L] + 22), wng16[L], 64, 1600, 1600);
    prep(P(cbase[L] + 18), wcl16[L], 64, 128, 128);
  }
  prep(P(50), wlast16, 16, 128, 128);

  auto cvt = [&](const float* s, _Float16* dp, int C, int dstCB, int coff) {
    int n = 8 * C * 512;
    k_convert<<<(n + 255) / 256, 256, 0, stream>>>(s, dp, C, dstCB, coff);
  };
  auto ln = [&](float* x, const float* g, const float* bt, int C) {
    k_layernorm<<<8, 256, 0, stream>>>(x, g, bt, C);
  };

  auto cell = [&](int L, const _Float16* x16, int Cin, int inCBx, int KpadX,
                  const float* kvp, int Y, int kvStride, float* h, float* c) {
    int Bv = cbase[L];
    // xcat = LN(conv(x))  (448 out ch -> 14 x 32-ch tiles)
    k_conv_wmma<25><<<dim3(256, 14), 32, 0, stream>>>(x16, wx16[L], P(Bv + 5), xcat, Cin, 448, KpadX, inCBx);
    ln(xcat, P(Bv + 17), P(Bv + 16), 448);
    // hcat = LN(conv(h))  (256 out ch -> 8 tiles)
    cvt(h, in16, 64, 64, 0);
    k_conv_wmma<25><<<dim3(256, 8), 32, 0, stream>>>(in16, wh16[L], P(Bv + 1), hcat, 64, 256, 1600, 64);
    ln(hcat, P(Bv + 9), P(Bv + 8), 256);
    // i_t, g_t, r_t
    k_gates1<<<1024, 256, 0, stream>>>(xcat, hcat, iT, gT, rT);
    // recall = eidetic_attn(r_t, c_hist)
    k_attn<<<dim3(32, 8), 32, 0, stream>>>(rT, kvp, recall, Y, kvStride);
    // c_new = i_t*g_t + LN(c + recall)
    k_add2<<<1024, 256, 0, stream>>>(c, recall, cn);
    ln(cn, P(Bv + 7), P(Bv + 6), 64);
    k_cnew<<<1024, 256, 0, stream>>>(iT, gT, cn, c);
    // gcat = LN(conv(gm)); gm_new
    cvt(gmem, in16, 64, 64, 0);
    k_conv_wmma<25><<<dim3(256, 8), 32, 0, stream>>>(in16, wm16[L], P(Bv + 2), gcat, 64, 256, 1600, 64);
    ln(gcat, P(Bv + 11), P(Bv + 10), 256);
    k_gates2<<<1024, 256, 0, stream>>>(xcat, gcat, gmem);
    // o_c, o_m, mem: pack [c_new | gm_new] as 128-ch f16 once, reuse 3x
    cvt(c, in16, 64, 128, 0);
    cvt(gmem, in16, 64, 128, 64);
    k_conv_wmma<25><<<dim3(256, 2), 32, 0, stream>>>(in16, wnc16[L], P(Bv + 3), oc, 64, 64, 1600, 128);
    ln(oc, P(Bv + 13), P(Bv + 12), 64);
    k_conv_wmma<25><<<dim3(256, 2), 32, 0, stream>>>(in16 + (64 << 9), wng16[L], P(Bv + 4), om, 64, 64, 1600, 128);
    ln(om, P(Bv + 15), P(Bv + 14), 64);
    k_conv_wmma<1><<<dim3(256, 2), 32, 0, stream>>>(in16, wcl16[L], P(Bv + 0), memb, 128, 64, 128, 128);
    // h_new = mem * tanh(tanh(o_x+o_h+o_c+o_m))
    k_outgate<<<1024, 256, 0, stream>>>(xcat, hcat, oc, om, memb, h);
  };

  for (int t = 0; t < 9; ++t) {
    // build sliding-window input (prev frame -> d0, new frame -> d1)
    k_frame_in<<<128, 256, 0, stream>>>(frames, maskp, genb, ci16, t);
    // layer 0: history = previous c0 (attention reads before c0 is overwritten)
    cell(0, ci16, 16, 16, 416, c0, 512, 32768, h0, c0);
    // layer 1: history = [zeros, c1@0, ..., c1@t] -> t+2 slots of 64 channels
    k_hist<<<1024, 256, 0, stream>>>(c1, hist1, t + 1);
    cvt(h0, in16, 64, 64, 0);
    cell(1, in16, 64, 64, 1600, hist1, 512 * (t + 2), 327680, h1, c1);
    // gen frame; last 5 steps also write the output slice
    cvt(h1, in16, 64, 64, 0);
    k_gen<<<128, 32, 0, stream>>>(in16, wlast16, genb,
                                  (t >= 4) ? (float*)d_out : nullptr, t - 4);
  }
}
